// MEBNETBlock_44693429682899
// MI455X (gfx1250) — matrix operand order repaired and checked by reading, not compiled, not run on hardware
//
#include <hip/hip_runtime.h>
#include <hip/hip_bf16.h>
#include <stdint.h>

typedef __attribute__((ext_vector_type(16))) __bf16 v16bf;
typedef __attribute__((ext_vector_type(8)))  float  v8f;

static constexpr int NN = 50000;   // nodes
static constexpr int NE = 800000;  // edges
static constexpr int NG = 512;     // graphs
static constexpr int H  = 128;

__device__ __forceinline__ float sspf(float x) {
  // shifted softplus: log(1+exp(x)) - log(2), overflow-safe
  float sp = (x > 20.0f) ? x : __logf(1.0f + __expf(x));
  return sp - 0.69314718055994530942f;
}

union Frag16 { v16bf v; uint4 q[2]; };

// C = ssp(A @ W + b)[ssp again if doubleSsp]
// A: [M,K] bf16, possibly a *virtual* concat of 128-wide segments (MODE).
// W: [K,N] bf16 row-major. B tile [K,64] staged in LDS once per block.
// Block = 256 threads = 8 wave32 waves; block tile 128x64; wave tile 16x64.
// K = NSEG*128 (constexpr) -> fully unrolled K loop, constexpr load offsets.
// MODE 0: contiguous A rows (stride K)
// MODE 1: edge concat  [A0[I0[m]], A0[I1[m]], A1[m], A2[I2[I0[m]]]]   (NSEG=4)
// MODE 2: triple concat [A0[m], A1[m], A2[I0?I0[m]:m]]                (NSEG=3)
// ZROW: 128 zero bf16 values; out-of-range rows read from it (no null branches).
template<int MODE, int NSEG>
__global__ __launch_bounds__(256)
void gemm_bias_ssp(const __bf16* __restrict__ A0, const __bf16* __restrict__ A1,
                   const __bf16* __restrict__ A2,
                   const int* __restrict__ I0, const int* __restrict__ I1,
                   const int* __restrict__ I2,
                   const __bf16* __restrict__ W, const float* __restrict__ bias,
                   const __bf16* __restrict__ ZROW,
                   __bf16* __restrict__ outBf, float* __restrict__ outF32,
                   const float* __restrict__ res,
                   int M, int N, int doubleSsp)
{
  constexpr int K = NSEG * 128;
  __shared__ __bf16 Bs[K * 64];

  const int tid    = threadIdx.x;
  const int lane   = tid & 31;
  const int wave   = tid >> 5;
  const int hi     = lane >> 4;
  const int tile_m = blockIdx.x * 128 + wave * 16;
  const int tile_n = blockIdx.y * 64;

  // ---- cooperative stage of W[:, tile_n:tile_n+64] into LDS ----
#pragma unroll
  for (int c = 0; c < K * 8 / 256; ++c) {
    const int idx = tid + c * 256;          // 16-byte chunk index
    const int k   = idx >> 2;               // 4 chunks per 64-wide row
    const int col = (idx & 3) * 16;
    *(uint4*)(&Bs[k * 64 + col]) =
        *(const uint4*)(W + (size_t)k * (size_t)N + tile_n + col);
  }
  __syncthreads();

  // ---- per-lane A segment pointers (gather fused here) ----
  const int  m    = tile_m + (lane & 15);
  const bool mval = (m < M);
  const __bf16* segs[NSEG];
  if (MODE == 0) {
    const __bf16* bb = mval ? (A0 + (size_t)m * (size_t)K) : ZROW;
    const size_t step = mval ? 128 : 0;
#pragma unroll
    for (int s = 0; s < NSEG; ++s) segs[s] = bb + (size_t)s * step;
  } else if (MODE == 1) {
    const int mc = mval ? m : 0;
    const int s0 = I0[mc], s1 = I1[mc];
    segs[0] = mval ? (A0 + (size_t)s0 * H)     : ZROW;  // n[src]
    segs[1] = mval ? (A0 + (size_t)s1 * H)     : ZROW;  // n[dst]
    segs[2] = mval ? (A1 + (size_t)m  * H)     : ZROW;  // e
    segs[3] = mval ? (A2 + (size_t)I2[s0] * H) : ZROW;  // g[batch[src]]
  } else {
    const int mc = mval ? m : 0;
    segs[0] = mval ? (A0 + (size_t)m * H) : ZROW;
    segs[1] = mval ? (A1 + (size_t)m * H) : ZROW;
    segs[2] = mval ? (A2 + (size_t)(I0 ? I0[mc] : mc) * H) : ZROW;
  }

  const __bf16* bq = &Bs[hi * 8 * 64 + (lane & 15)];   // B fragment base: lane col n = lane&15; k = k0+hi*8+s, k0+16+hi*8+(s-8)

  v8f acc[4] = {};
#pragma unroll
  for (int k0 = 0; k0 < K; k0 += 32) {
    // A fragment: lane row m; halves 0..7 at k0+hi*8, 8..15 at k0+16+hi*8
    Frag16 a;
    const __bf16* sp = segs[k0 >> 7] + (k0 & 127) + hi * 8;
    a.q[0] = *(const uint4*)(sp);
    a.q[1] = *(const uint4*)(sp + 16);
#pragma unroll
    for (int j = 0; j < 4; ++j) {
      Frag16 b;
#pragma unroll
      for (int s = 0; s < 8; ++s) {
        b.v[s]     = bq[(k0 + s) * 64 + j * 16];
        b.v[8 + s] = bq[(k0 + 16 + s) * 64 + j * 16];
      }
      acc[j] = __builtin_amdgcn_wmma_f32_16x16x32_bf16(
          false, a.v, false, b.v, (short)0, acc[j], false, false);
    }
  }

  // C/D layout: lane -> col tile_n+16j+(lane&15); VGPR r -> row tile_m + hi*8 + r
#pragma unroll
  for (int j = 0; j < 4; ++j) {
    const int n  = tile_n + j * 16 + (lane & 15);
    const float bv = bias[n];
#pragma unroll
    for (int r = 0; r < 8; ++r) {
      const int mm = tile_m + hi * 8 + r;
      if (mm < M) {
        float v   = acc[j][r] + bv;
        float act = sspf(v);
        if (doubleSsp) act = sspf(act);
        const size_t o = (size_t)mm * (size_t)N + n;
        if (outBf)  outBf[o]  = (__bf16)act;
        if (outF32) outF32[o] = act + (res ? res[o] : 0.0f);
      }
    }
  }
}

__global__ void cvt_f32_bf16(const float* __restrict__ src, __bf16* __restrict__ dst,
                             size_t n4)   // n4 = count of float4 groups
{
  size_t i = (size_t)blockIdx.x * blockDim.x + threadIdx.x;
  if (i < n4) {
    const float4 f = *(const float4*)(src + i * 4);
    __bf16 o[4] = {(__bf16)f.x, (__bf16)f.y, (__bf16)f.z, (__bf16)f.w};
    *(uint2*)(dst + i * 4) = *(const uint2*)o;
  }
}

// One wave per row: scatter-add bf16 row [H] into f32 accumulator + count.
// segment = idxB ? idxB[idxA[i]] : idxA[i]
__global__ void scatter_add_bf16(const __bf16* __restrict__ vals,
                                 const int* __restrict__ idxA,
                                 const int* __restrict__ idxB,
                                 float* __restrict__ acc, float* __restrict__ cnt,
                                 int M)
{
  int gw     = (int)((blockIdx.x * blockDim.x + threadIdx.x) >> 5);
  int lane   = threadIdx.x & 31;
  int stride = (int)((gridDim.x * (size_t)blockDim.x) >> 5);
  for (int i = gw; i < M; i += stride) {
    int s = idxA[i];
    if (idxB) s = idxB[s];
    const uint2 u = *(const uint2*)(vals + (size_t)i * H + lane * 4);
    float* a = acc + (size_t)s * H + lane * 4;
    atomicAdd(a + 0, __uint_as_float(u.x << 16));
    atomicAdd(a + 1, __uint_as_float(u.x & 0xffff0000u));
    atomicAdd(a + 2, __uint_as_float(u.y << 16));
    atomicAdd(a + 3, __uint_as_float(u.y & 0xffff0000u));
    if (lane == 0) atomicAdd(cnt + s, 1.0f);
  }
}

__global__ void seg_normalize(const float* __restrict__ acc, const float* __restrict__ cnt,
                              __bf16* __restrict__ out, int S)
{
  int i = blockIdx.x * blockDim.x + threadIdx.x;
  if (i < S * H) out[i] = (__bf16)(acc[i] / fmaxf(cnt[i >> 7], 1.0f));
}

extern "C" void kernel_launch(void* const* d_in, const int* in_sizes, int n_in,
                              void* d_out, int out_size, void* d_ws, size_t ws_size,
                              hipStream_t stream)
{
  (void)in_sizes; (void)n_in; (void)out_size; (void)ws_size;

  const float* node_feats = (const float*)d_in[0];
  const float* edge_feats = (const float*)d_in[1];
  const float* glob_feats = (const float*)d_in[2];
  const int*   eidx       = (const int*)d_in[3];
  const int*   src        = eidx;
  const int*   dst        = eidx + NE;
  const int*   batch      = (const int*)d_in[4];

  // params pytree-flattened in sorted-key order: em, gm, le1, lg1, ln1, nm;
  // each layer contributes (W, b) pairs in order.
  const float* emW[3]  = {(const float*)d_in[5],  (const float*)d_in[7],  (const float*)d_in[9]};
  const float* emB[3]  = {(const float*)d_in[6],  (const float*)d_in[8],  (const float*)d_in[10]};
  const float* gmW[3]  = {(const float*)d_in[11], (const float*)d_in[13], (const float*)d_in[15]};
  const float* gmB[3]  = {(const float*)d_in[12], (const float*)d_in[14], (const float*)d_in[16]};
  const float* le1W[2] = {(const float*)d_in[17], (const float*)d_in[19]};
  const float* le1B[2] = {(const float*)d_in[18], (const float*)d_in[20]};
  const float* lg1W[2] = {(const float*)d_in[21], (const float*)d_in[23]};
  const float* lg1B[2] = {(const float*)d_in[22], (const float*)d_in[24]};
  const float* ln1W[2] = {(const float*)d_in[25], (const float*)d_in[27]};
  const float* ln1B[2] = {(const float*)d_in[26], (const float*)d_in[28]};
  const float* nmW[3]  = {(const float*)d_in[29], (const float*)d_in[31], (const float*)d_in[33]};
  const float* nmB[3]  = {(const float*)d_in[30], (const float*)d_in[32], (const float*)d_in[34]};

  float* out_n = (float*)d_out;
  float* out_e = out_n + (size_t)NN * H;
  float* out_g = out_e + (size_t)NE * H;

  // ---- workspace bump allocator ----
  char* base = (char*)d_ws;
  size_t off = 0;
  auto bump = [&](size_t bytes) -> void* {
    size_t a = (off + 255) & ~(size_t)255;
    off = a + bytes;
    return base + a;
  };

  // bf16 weights, kept in natural [K,N] layout
  struct WSpec { const float* w; int k, n; __bf16* bf; };
  WSpec ws[15] = {
    {emW[0], 512, 256, 0}, {emW[1], 256, 256, 0}, {emW[2], 256, 128, 0},
    {gmW[0], 384, 256, 0}, {gmW[1], 256, 256, 0}, {gmW[2], 256, 128, 0},
    {le1W[0], 128, 256, 0}, {le1W[1], 256, 128, 0},
    {lg1W[0], 128, 256, 0}, {lg1W[1], 256, 128, 0},
    {ln1W[0], 128, 256, 0}, {ln1W[1], 256, 128, 0},
    {nmW[0], 384, 256, 0}, {nmW[1], 256, 256, 0}, {nmW[2], 256, 128, 0},
  };
  for (int i = 0; i < 15; ++i)
    ws[i].bf = (__bf16*)bump((size_t)ws[i].k * ws[i].n * sizeof(__bf16));

  __bf16* ZROW = (__bf16*)bump((size_t)H * 2);        // zero row for OOR tiles
  __bf16* EF   = (__bf16*)bump((size_t)NE * H * 2);   // edge feats bf16; reused as e_new
  __bf16* NF   = (__bf16*)bump((size_t)NN * H * 2);   // node feats bf16; reused as n_new
  __bf16* GF   = (__bf16*)bump((size_t)NG * H * 2);
  __bf16* ELN  = (__bf16*)bump((size_t)NE * H * 2);   // e after le1
  __bf16* NLN  = (__bf16*)bump((size_t)NN * H * 2);   // n after ln1
  __bf16* GLN  = (__bf16*)bump((size_t)NG * H * 2);   // g after lg1
  __bf16* HB1  = (__bf16*)bump((size_t)NE * 2 * H * 2); // 2H hidden ping
  __bf16* HB2  = (__bf16*)bump((size_t)NE * 2 * H * 2); // 2H hidden pong
  float*  AGEN = (float*)bump((size_t)NN * H * 4);    // agg e->node
  float*  CNTN = (float*)bump((size_t)NN * 4);
  __bf16* AGENB = (__bf16*)bump((size_t)NN * H * 2);
  float*  AGEG = (float*)bump((size_t)NG * H * 4);    // agg e->graph
  float*  CNTEG = (float*)bump((size_t)NG * 4);
  __bf16* AGEGB = (__bf16*)bump((size_t)NG * H * 2);
  float*  AGNG = (float*)bump((size_t)NG * H * 4);    // agg n->graph
  float*  CNTNG = (float*)bump((size_t)NG * 4);
  __bf16* AGNGB = (__bf16*)bump((size_t)NG * H * 2);
  __bf16* ENEW = EF;   // aliases: source bf16 copies are dead by then
  __bf16* NNEW = NF;

  // zero the zero-row and accumulators every call (deterministic)
  hipMemsetAsync(ZROW,  0, (size_t)H * 2, stream);
  hipMemsetAsync(AGEN,  0, (size_t)NN * H * 4, stream);
  hipMemsetAsync(CNTN,  0, (size_t)NN * 4, stream);
  hipMemsetAsync(AGEG,  0, (size_t)NG * H * 4, stream);
  hipMemsetAsync(CNTEG, 0, (size_t)NG * 4, stream);
  hipMemsetAsync(AGNG,  0, (size_t)NG * H * 4, stream);
  hipMemsetAsync(CNTNG, 0, (size_t)NG * 4, stream);

  auto cvt = [&](const float* s, __bf16* d, size_t n) {
    size_t n4 = n / 4;
    cvt_f32_bf16<<<(unsigned)((n4 + 255) / 256), 256, 0, stream>>>(s, d, n4);
  };
  auto gemm = [&](int mode, int nseg, const __bf16* A0, const __bf16* A1,
                  const __bf16* A2, const int* i0, const int* i1, const int* i2,
                  const __bf16* Wb, const float* b,
                  __bf16* ob, float* of, const float* r,
                  int M, int N, int dssp) {
    dim3 g((unsigned)((M + 127) / 128), (unsigned)(N / 64)), blk(256);
    if (mode == 0 && nseg == 1)
      gemm_bias_ssp<0, 1><<<g, blk, 0, stream>>>(A0, A1, A2, i0, i1, i2, Wb, b, ZROW, ob, of, r, M, N, dssp);
    else if (mode == 0 && nseg == 2)
      gemm_bias_ssp<0, 2><<<g, blk, 0, stream>>>(A0, A1, A2, i0, i1, i2, Wb, b, ZROW, ob, of, r, M, N, dssp);
    else if (mode == 1)
      gemm_bias_ssp<1, 4><<<g, blk, 0, stream>>>(A0, A1, A2, i0, i1, i2, Wb, b, ZROW, ob, of, r, M, N, dssp);
    else
      gemm_bias_ssp<2, 3><<<g, blk, 0, stream>>>(A0, A1, A2, i0, i1, i2, Wb, b, ZROW, ob, of, r, M, N, dssp);
  };
  auto scat = [&](const __bf16* v, const int* ia, const int* ib, float* acc, float* c, int M) {
    scatter_add_bf16<<<(unsigned)((M + 7) / 8), 256, 0, stream>>>(v, ia, ib, acc, c, M);
  };
  auto norm = [&](const float* acc, const float* c, __bf16* o, int S) {
    seg_normalize<<<(unsigned)((S * H + 255) / 256), 256, 0, stream>>>(acc, c, o, S);
  };

  // ---- weight + input conversion to bf16 ----
  for (int i = 0; i < 15; ++i) cvt(ws[i].w, ws[i].bf, (size_t)ws[i].k * ws[i].n);
  cvt(node_feats, NF, (size_t)NN * H);
  cvt(edge_feats, EF, (size_t)NE * H);
  cvt(glob_feats, GF, (size_t)NG * H);

  // ---- input transforms: x = ssp(mlp(x)); mlp already ends in ssp -> doubleSsp on layer 2
  gemm(0, 1, NF, 0, 0, 0, 0, 0, ws[10].bf, ln1B[0], HB1, 0, 0, NN, 256, 0);
  gemm(0, 2, HB1, 0, 0, 0, 0, 0, ws[11].bf, ln1B[1], NLN, 0, 0, NN, 128, 1);
  gemm(0, 1, EF, 0, 0, 0, 0, 0, ws[6].bf, le1B[0], HB1, 0, 0, NE, 256, 0);
  gemm(0, 2, HB1, 0, 0, 0, 0, 0, ws[7].bf, le1B[1], ELN, 0, 0, NE, 128, 1);
  gemm(0, 1, GF, 0, 0, 0, 0, 0, ws[8].bf, lg1B[0], HB1, 0, 0, NG, 256, 0);
  gemm(0, 2, HB1, 0, 0, 0, 0, 0, ws[9].bf, lg1B[1], GLN, 0, 0, NG, 128, 1);

  // ---- EdgeModel: virtual concat [n[src], n[dst], e, g[batch[src]]] -> 4H->2H->2H->H
  gemm(1, 4, NLN, ELN, GLN, src, dst, batch, ws[0].bf, emB[0], HB1, 0, 0, NE, 256, 0);
  gemm(0, 2, HB1, 0, 0, 0, 0, 0, ws[1].bf, emB[1], HB2, 0, 0, NE, 256, 0);
  gemm(0, 2, HB2, 0, 0, 0, 0, 0, ws[2].bf, emB[2], ENEW, out_e, edge_feats, NE, 128, 0);

  // ---- segment means of e_new
  scat(ENEW, dst, 0, AGEN, CNTN, NE);          // mean over dst nodes
  norm(AGEN, CNTN, AGENB, NN);
  scat(ENEW, src, batch, AGEG, CNTEG, NE);     // mean over batch[src] graphs
  norm(AGEG, CNTEG, AGEGB, NG);

  // ---- NodeModel: concat [n, agg_e_node, g[batch]] -> 3H->2H->2H->H
  gemm(2, 3, NLN, AGENB, GLN, batch, 0, 0, ws[12].bf, nmB[0], HB1, 0, 0, NN, 256, 0);
  gemm(0, 2, HB1, 0, 0, 0, 0, 0, ws[13].bf, nmB[1], HB2, 0, 0, NN, 256, 0);
  gemm(0, 2, HB2, 0, 0, 0, 0, 0, ws[14].bf, nmB[2], NNEW, out_n, node_feats, NN, 128, 0);

  // ---- mean of n_new over graphs
  scat(NNEW, batch, 0, AGNG, CNTNG, NN);
  norm(AGNG, CNTNG, AGNGB, NG);

  // ---- GlobalModel: concat [agg_n_g, agg_e_g, g] -> 3H->2H->2H->H
  gemm(2, 3, AGNGB, AGEGB, GLN, 0, 0, 0, ws[3].bf, gmB[0], HB1, 0, 0, NG, 256, 0);
  gemm(0, 2, HB1, 0, 0, 0, 0, 0, ws[4].bf, gmB[1], HB2, 0, 0, NG, 256, 0);
  gemm(0, 2, HB2, 0, 0, 0, 0, 0, ws[5].bf, gmB[2], (__bf16*)0, out_g, glob_feats, NG, 128, 0);
}
